// CamadaLinear_22333829939896
// MI455X (gfx1250) — matrix operand order repaired and checked by reading, not compiled, not run on hardware
//
#include <hip/hip_runtime.h>

// GEMM: Y[M,N] = X[M,K] @ (peso * block_scale)^T
// M = B*S = 4096, K = d_in = 4096, N = d_out = 12288, scale blocks 128x128.
#define M_TOT 4096
#define N_TOT 12288
#define K_TOT 4096
#define BM 128
#define BN 256
#define BK 32
#define LDST 40            // padded LDS row stride (bf16 elems): 80B rows -> conflict-free b128 frag loads
#define NKT (K_TOT / BK)   // 128 K-steps
#define SBLK (K_TOT / 128) // 32 scale columns

typedef __attribute__((ext_vector_type(16))) __bf16 v16bf;
typedef __attribute__((ext_vector_type(8)))  __bf16 v8bf;
typedef __attribute__((ext_vector_type(8)))  float  v8f;

union Frag16 { v16bf v; v8bf h[2]; };

// two f32 -> packed 2x bf16 (truncate) in ONE v_perm_b32:
// result = (a >> 16) | (b & 0xffff0000)
__device__ __forceinline__ unsigned pack_bf16x2(float a, float b) {
    return __builtin_amdgcn_perm(__float_as_uint(b), __float_as_uint(a), 0x07060302u);
}

__device__ __forceinline__ uint4 pack_f8(const float4& p, const float4& q) {
    uint4 r;
    r.x = pack_bf16x2(p.x, p.y);
    r.y = pack_bf16x2(p.z, p.w);
    r.z = pack_bf16x2(q.x, q.y);
    r.w = pack_bf16x2(q.z, q.w);
    return r;
}

__device__ __forceinline__ float4 scale4(const float4& p, float s) {
    return make_float4(p.x * s, p.y * s, p.z * s, p.w * s);
}

__global__ __launch_bounds__(256, 1)
void camada_linear_bf16_wmma(const float* __restrict__ X,
                             const float* __restrict__ Wq,   // peso [N,K]
                             const float* __restrict__ Sc,   // escala [N/128, K/128]
                             float* __restrict__ Y) {
    __shared__ __bf16 sx[2][BM * LDST];   // 20 KB
    __shared__ __bf16 sw[2][BN * LDST];   // 40 KB

    const int tid  = threadIdx.x;
    const int lane = tid & 31;
    const int wave = tid >> 5;
    const int wm   = wave & 1;    // M sub-tile: 64 rows
    const int wn   = wave >> 1;   // N sub-tile: 64 cols
    const int hl   = lane >> 4;   // lane half
    const int r    = lane & 15;

    const int nBase = blockIdx.x * BN;
    const int mBase = blockIdx.y * BM;

    // X staging: 2 threads per row, 16 f32 each (4x b128)
    const int ldRow = tid >> 1;
    const int ldCol = (tid & 1) * 16;
    const float* xg = X  + (size_t)(mBase + ldRow) * K_TOT + ldCol;
    // W staging: 1 thread per row (256 rows), 32 f32 each (8x b128)
    const float* wg = Wq + (size_t)(nBase + tid) * K_TOT;

    // scale row select: rows [0,128) -> escala row 2*bx, rows [128,256) -> 2*bx+1
    const float* scA = Sc + (size_t)(2 * blockIdx.x) * SBLK;
    const float* scB = scA + SBLK;
    const bool loRow = (tid < 128);

    v8f acc[4][4];
#pragma unroll
    for (int i = 0; i < 4; ++i)
#pragma unroll
        for (int j = 0; j < 4; ++j)
            acc[i][j] = v8f{};

    float4 xr[4], wr[8];

    // ---- prologue: tile 0 -> regs -> LDS buf 0 (scale folded into weights)
#pragma unroll
    for (int j = 0; j < 4; ++j) xr[j] = *(const float4*)(xg + 4 * j);
#pragma unroll
    for (int j = 0; j < 8; ++j) wr[j] = *(const float4*)(wg + 4 * j);
    {
        const float s = loRow ? scA[0] : scB[0];
        __bf16* dx = &sx[0][ldRow * LDST + ldCol];
        __bf16* dw = &sw[0][tid * LDST];
        *(uint4*)(dx)      = pack_f8(xr[0], xr[1]);
        *(uint4*)(dx + 8)  = pack_f8(xr[2], xr[3]);
        *(uint4*)(dw)      = pack_f8(scale4(wr[0], s), scale4(wr[1], s));
        *(uint4*)(dw + 8)  = pack_f8(scale4(wr[2], s), scale4(wr[3], s));
        *(uint4*)(dw + 16) = pack_f8(scale4(wr[4], s), scale4(wr[5], s));
        *(uint4*)(dw + 24) = pack_f8(scale4(wr[6], s), scale4(wr[7], s));
    }
    __syncthreads();

    // ---- main K loop, double-buffered software pipeline
    for (int kb = 0; kb < NKT; ++kb) {
        const int buf = kb & 1;

        if (kb + 1 < NKT) {
            const float* xgn = xg + (size_t)(kb + 1) * BK;
            const float* wgn = wg + (size_t)(kb + 1) * BK;
#pragma unroll
            for (int j = 0; j < 4; ++j) xr[j] = *(const float4*)(xgn + 4 * j);
#pragma unroll
            for (int j = 0; j < 8; ++j) wr[j] = *(const float4*)(wgn + 4 * j);
            if (kb + 2 < NKT) {   // L2 prefetch hints (global_prefetch_b8)
                __builtin_prefetch(xg + (size_t)(kb + 2) * BK, 0, 1);
                __builtin_prefetch(wg + (size_t)(kb + 2) * BK, 0, 1);
            }
        }

        // fragments + WMMA
        const __bf16* sxb = sx[buf];
        const __bf16* swb = sw[buf];

        Frag16 bfr[4];
#pragma unroll
        for (int tj = 0; tj < 4; ++tj) {
            // B (K x N): lane = column n; lanes 0-15 hold K0..15, lanes 16-31 hold K16..31
            const __bf16* p = swb + (wn * 64 + tj * 16 + r) * LDST + hl * 16;
            bfr[tj].h[0] = *(const v8bf*)(p);
            bfr[tj].h[1] = *(const v8bf*)(p + 8);
        }
#pragma unroll
        for (int ti = 0; ti < 4; ++ti) {
            // A (M x K): lane = row m; half 0: K0..15, half 1: K16..31
            Frag16 a;
            const __bf16* p = sxb + (wm * 64 + ti * 16 + r) * LDST;
            a.h[0] = *(const v8bf*)(p + hl * 16);
            a.h[1] = *(const v8bf*)(p + hl * 16 + 8);
#pragma unroll
            for (int tj = 0; tj < 4; ++tj) {
                acc[ti][tj] = __builtin_amdgcn_wmma_f32_16x16x32_bf16(
                    false, a.v, false, bfr[tj].v, (short)0, acc[ti][tj],
                    false, false);
            }
        }

        if (kb + 1 < NKT) {
            const int k4 = (kb + 1) >> 2;
            const float s = loRow ? scA[k4] : scB[k4];
            const int nb = (kb + 1) & 1;
            __bf16* dx = &sx[nb][ldRow * LDST + ldCol];
            __bf16* dw = &sw[nb][tid * LDST];
            *(uint4*)(dx)      = pack_f8(xr[0], xr[1]);
            *(uint4*)(dx + 8)  = pack_f8(xr[2], xr[3]);
            *(uint4*)(dw)      = pack_f8(scale4(wr[0], s), scale4(wr[1], s));
            *(uint4*)(dw + 8)  = pack_f8(scale4(wr[2], s), scale4(wr[3], s));
            *(uint4*)(dw + 16) = pack_f8(scale4(wr[4], s), scale4(wr[5], s));
            *(uint4*)(dw + 24) = pack_f8(scale4(wr[6], s), scale4(wr[7], s));
            __syncthreads();
        }
    }

    // ---- epilogue: 16x16 f32 D layout -> VGPR v holds M=v (lanes 0-15) / M=v+8 (lanes 16-31)
    float* yg = Y + (size_t)mBase * N_TOT + nBase;
#pragma unroll
    for (int ti = 0; ti < 4; ++ti) {
#pragma unroll
        for (int tj = 0; tj < 4; ++tj) {
            const int col = wn * 64 + tj * 16 + r;
#pragma unroll
            for (int v = 0; v < 8; ++v) {
                const int rowm = wm * 64 + ti * 16 + v + 8 * hl;
                yg[(size_t)rowm * N_TOT + col] = acc[ti][tj][v];
            }
        }
    }
}

extern "C" void kernel_launch(void* const* d_in, const int* in_sizes, int n_in,
                              void* d_out, int out_size, void* d_ws, size_t ws_size,
                              hipStream_t stream) {
    (void)n_in; (void)d_ws; (void)ws_size; (void)out_size; (void)in_sizes;
    const float* x      = (const float*)d_in[0];   // [B,S,d_in]  f32
    const float* peso   = (const float*)d_in[1];   // [d_out,d_in] f32 (fp8-valued)
    const float* escala = (const float*)d_in[2];   // [d_out/128, d_in/128] f32
    float* y = (float*)d_out;                      // [B,S,d_out] f32

    dim3 grid(N_TOT / BN, M_TOT / BM, 1);          // 48 x 32 workgroups
    dim3 block(256, 1, 1);                         // 8 wave32 waves
    camada_linear_bf16_wmma<<<grid, block, 0, stream>>>(x, peso, escala, y);
}
